// MDAGCN_block_26113401159749
// MI455X (gfx1250) — matrix operand order repaired and checked by reading, not compiled, not run on hardware
//
#include <hip/hip_runtime.h>

// ---------------- problem constants ----------------
#define N_    8
#define T_    8
#define V_    128
#define FD_   64
#define C1_   256
#define CHEV_ 64
#define TF_   64
#define KT_   3
#define NT_   (N_*T_)
#define ALPHA_ 1e-4f
#define YSZ_  (N_*T_*V_*TF_)   // 524288

typedef __attribute__((ext_vector_type(16))) _Float16 v16h;
typedef __attribute__((ext_vector_type(8)))  float    v8f;

// ================= WMMA fragment helpers (gfx1250, wave32) =================
// A (16x32, f16): lanes 0-15 -> M=lane, K slots {0..7,16..23}; lanes 16-31 ->
// M=lane-16, K slots {8..15,24..31}.  Element e: k = e + (e>=8?8:0) + 8*half.
// Per lane these are two contiguous 8-float runs -> vectorized b128 loads.
__device__ __forceinline__ v16h load_a_frag_vec(const float* __restrict__ A, int lda,
                                                int m0, int k0) {
    int lane = threadIdx.x & 31;
    int half = lane >> 4;
    int m = m0 + (lane & 15);
    const float* p = A + (long)m * lda + k0 + half * 8;
    float4 r0 = *(const float4*)(p);
    float4 r1 = *(const float4*)(p + 4);
    float4 r2 = *(const float4*)(p + 16);
    float4 r3 = *(const float4*)(p + 20);
    v16h a;
    a[0]  = (_Float16)r0.x; a[1]  = (_Float16)r0.y; a[2]  = (_Float16)r0.z; a[3]  = (_Float16)r0.w;
    a[4]  = (_Float16)r1.x; a[5]  = (_Float16)r1.y; a[6]  = (_Float16)r1.z; a[7]  = (_Float16)r1.w;
    a[8]  = (_Float16)r2.x; a[9]  = (_Float16)r2.y; a[10] = (_Float16)r2.z; a[11] = (_Float16)r2.w;
    a[12] = (_Float16)r3.x; a[13] = (_Float16)r3.y; a[14] = (_Float16)r3.z; a[15] = (_Float16)r3.w;
    return a;
}
// B (32x16, f16): lanes 0-15 -> N=lane, K {0..7,16..23}; lanes 16-31 -> N=lane-16, K {8..15,24..31}.
__device__ __forceinline__ v16h load_b_frag(const float* __restrict__ B, int ldb,
                                            int k0, int n0) {
    int lane = threadIdx.x & 31;
    int half = lane >> 4;
    int n = n0 + (lane & 15);
    v16h b;
#pragma unroll
    for (int e = 0; e < 16; ++e) {
        int k = k0 + e + ((e >= 8) ? 8 : 0) + half * 8;
        b[e] = (_Float16)B[(long)k * ldb + n];
    }
    return b;
}
// C/D (16x16 f32, 8 VGPRs): VGPR r -> row m0+r (lanes0-15) / m0+r+8 (lanes16-31), col lane&15
__device__ __forceinline__ v8f load_c_frag(const float* __restrict__ C, int ldc,
                                           int m0, int n0) {
    int lane = threadIdx.x & 31;
    int half = lane >> 4;
    int n = n0 + (lane & 15);
    v8f c;
#pragma unroll
    for (int r = 0; r < 8; ++r) c[r] = C[(long)(m0 + r + half * 8) * ldc + n];
    return c;
}
__device__ __forceinline__ void store_c_frag(float* __restrict__ C, int ldc,
                                             int m0, int n0, v8f c) {
    int lane = threadIdx.x & 31;
    int half = lane >> 4;
    int n = n0 + (lane & 15);
#pragma unroll
    for (int r = 0; r < 8; ++r) C[(long)(m0 + r + half * 8) * ldc + n] = c[r];
}

// ============ generic batched GEMM:  C[b] (+)= A[b] (MxK) * B[b] (KxN) ============
// one wave per 16x64 macro-tile (A-frag reused across 4 WMMAs);
// grid = (batch, M/16, N/64), block = 32; K, N compile-time -> fully unrolled
template <int Kd, int Nn>
__global__ __launch_bounds__(32)
void wmma_gemm_batched(const float* __restrict__ A, const float* __restrict__ B,
                       float* __restrict__ C, long sA, long sB, long sC, int init) {
    int batch = blockIdx.x;
    int m0 = blockIdx.y * 16, n0 = blockIdx.z * 64;
    const float* Ab = A + (long)batch * sA;
    const float* Bb = B + (long)batch * sB;
    float* Cb = C + (long)batch * sC;
    v8f c0 = {}, c1 = {}, c2 = {}, c3 = {};
    if (!init) {
        c0 = load_c_frag(Cb, Nn, m0, n0);
        c1 = load_c_frag(Cb, Nn, m0, n0 + 16);
        c2 = load_c_frag(Cb, Nn, m0, n0 + 32);
        c3 = load_c_frag(Cb, Nn, m0, n0 + 48);
    }
#pragma unroll
    for (int k0 = 0; k0 < Kd; k0 += 32) {
        v16h a = load_a_frag_vec(Ab, Kd, m0, k0);
        v16h b0 = load_b_frag(Bb, Nn, k0, n0);
        c0 = __builtin_amdgcn_wmma_f32_16x16x32_f16(false, a, false, b0, (short)0, c0, false, false);
        v16h b1 = load_b_frag(Bb, Nn, k0, n0 + 16);
        c1 = __builtin_amdgcn_wmma_f32_16x16x32_f16(false, a, false, b1, (short)0, c1, false, false);
        v16h b2 = load_b_frag(Bb, Nn, k0, n0 + 32);
        c2 = __builtin_amdgcn_wmma_f32_16x16x32_f16(false, a, false, b2, (short)0, c2, false, false);
        v16h b3 = load_b_frag(Bb, Nn, k0, n0 + 48);
        c3 = __builtin_amdgcn_wmma_f32_16x16x32_f16(false, a, false, b3, (short)0, c3, false, false);
    }
    store_c_frag(Cb, Nn, m0, n0, c0);
    store_c_frag(Cb, Nn, m0, n0 + 16, c1);
    store_c_frag(Cb, Nn, m0, n0 + 32, c2);
    store_c_frag(Cb, Nn, m0, n0 + 48, c3);
}

// ===== Cheb aggregation: G[nt] = (P_k(L) . Att)^T @ X   (A built on the fly) =====
// Aeff[i][j] = poly(L[nt][j][i]) * Att[n][j][i];  poly k=1: L,  k=2: 2*L*L - eye
// grid = (nt, V/16, Fin/64), block = 32
template <int Fin>
__global__ __launch_bounds__(32)
void cheb_agg_wmma(const float* __restrict__ L, const float* __restrict__ Att,
                   const float* __restrict__ X, float* __restrict__ G, int kpoly) {
    int nt = blockIdx.x;
    int n = nt >> 3;
    int m0 = blockIdx.y * 16, n0 = blockIdx.z * 64;
    const float* Lb = L + (long)nt * V_ * V_;
    const float* Ab = Att + (long)n * V_ * V_;
    const float* Xb = X + (long)nt * V_ * Fin;
    float* Gb = G + (long)nt * V_ * Fin;
    int lane = threadIdx.x & 31;
    int half = lane >> 4;
    int m = m0 + (lane & 15);
    v8f c0 = {}, c1 = {}, c2 = {}, c3 = {};
#pragma unroll
    for (int k0 = 0; k0 < V_; k0 += 32) {
        v16h a;
#pragma unroll
        for (int e = 0; e < 16; ++e) {
            int j = k0 + e + ((e >= 8) ? 8 : 0) + half * 8;
            float l = Lb[(long)j * V_ + m];
            float p = (kpoly == 1) ? l : (2.f * l * l - ((j == m) ? 1.f : 0.f));
            a[e] = (_Float16)(p * Ab[(long)j * V_ + m]);
        }
        v16h b0 = load_b_frag(Xb, Fin, k0, n0);
        c0 = __builtin_amdgcn_wmma_f32_16x16x32_f16(false, a, false, b0, (short)0, c0, false, false);
        v16h b1 = load_b_frag(Xb, Fin, k0, n0 + 16);
        c1 = __builtin_amdgcn_wmma_f32_16x16x32_f16(false, a, false, b1, (short)0, c1, false, false);
        v16h b2 = load_b_frag(Xb, Fin, k0, n0 + 32);
        c2 = __builtin_amdgcn_wmma_f32_16x16x32_f16(false, a, false, b2, (short)0, c2, false, false);
        v16h b3 = load_b_frag(Xb, Fin, k0, n0 + 48);
        c3 = __builtin_amdgcn_wmma_f32_16x16x32_f16(false, a, false, b3, (short)0, c3, false, false);
    }
    store_c_frag(Gb, Fin, m0, n0, c0);
    store_c_frag(Gb, Fin, m0, n0 + 16, c1);
    store_c_frag(Gb, Fin, m0, n0 + 32, c2);
    store_c_frag(Gb, Fin, m0, n0 + 48, c3);
}

// ================= temporal attention: TAtt (N,T,T), softmax over axis=1 =================
__global__ void k_tatt(const float* __restrict__ x, const float* __restrict__ U1,
                       const float* __restrict__ U2, const float* __restrict__ U3,
                       const float* __restrict__ b_e, const float* __restrict__ V_e,
                       float* __restrict__ TAtt) {
    int n = blockIdx.x;
    int tid = threadIdx.x; // 128
    __shared__ float tmp[T_][FD_];
    __shared__ float lhs[T_][V_];
    __shared__ float rhs[V_][T_];
    __shared__ float sg[T_][T_];
    __shared__ float Sm[T_][T_];
    const float* xb = x + (long)n * T_ * V_ * FD_;
    for (int idx = tid; idx < T_ * FD_; idx += 128) {
        int t = idx / FD_, f = idx % FD_;
        float s = 0.f;
        for (int v = 0; v < V_; ++v) s += xb[((long)t * V_ + v) * FD_ + f] * U1[v];
        tmp[t][f] = s;
    }
    __syncthreads();
    for (int idx = tid; idx < T_ * V_; idx += 128) {
        int t = idx / V_, v = idx % V_;
        float s = 0.f;
        for (int f = 0; f < FD_; ++f) s += tmp[t][f] * U2[f * V_ + v];
        lhs[t][v] = s;
    }
    for (int idx = tid; idx < V_ * T_; idx += 128) {
        int v = idx / T_, t = idx % T_;
        float s = 0.f;
        for (int f = 0; f < FD_; ++f) s += xb[((long)t * V_ + v) * FD_ + f] * U3[f];
        rhs[v][t] = s;
    }
    __syncthreads();
    if (tid < T_ * T_) {
        int t = tid / T_, s2 = tid % T_;
        float s = 0.f;
        for (int v = 0; v < V_; ++v) s += lhs[t][v] * rhs[v][s2];
        s += b_e[t * T_ + s2];
        sg[t][s2] = 1.f / (1.f + expf(-s));
    }
    __syncthreads();
    if (tid < T_ * T_) {
        int i = tid / T_, j = tid % T_;
        float s = 0.f;
        for (int k = 0; k < T_; ++k) s += V_e[i * T_ + k] * sg[k][j];
        Sm[i][j] = s;
    }
    __syncthreads();
    if (tid < T_) {
        int j = tid;
        float mx = -1e30f;
        for (int i = 0; i < T_; ++i) mx = fmaxf(mx, Sm[i][j]);
        float e[T_], ssum = 0.f;
        for (int i = 0; i < T_; ++i) { e[i] = expf(Sm[i][j] - mx); ssum += e[i]; }
        for (int i = 0; i < T_; ++i) TAtt[(n * T_ + i) * T_ + j] = e[i] / ssum;
    }
}

// x_TAtt[n,s,v,f] = sum_t x[n,t,v,f] * TAtt[n,t,s]
__global__ void k_xtatt(const float* __restrict__ x, const float* __restrict__ TAtt,
                        float* __restrict__ xT) {
    long idx = (long)blockIdx.x * 256 + threadIdx.x;
    int f = idx & (FD_ - 1);
    long r = idx >> 6;
    int v = r & (V_ - 1); r >>= 7;
    int s2 = r & (T_ - 1);
    int n = (int)(r >> 3);
    float acc = 0.f;
    for (int t = 0; t < T_; ++t)
        acc += x[(((long)(n * T_ + t) * V_) + v) * FD_ + f] * TAtt[(n * T_ + t) * T_ + s2];
    xT[idx] = acc;
}

// spatial attention prelude -> Psig[n,v,u] = sigmoid(prod + b_s)
__global__ void k_satt_pre(const float* __restrict__ xT, const float* __restrict__ W1,
                           const float* __restrict__ W2, const float* __restrict__ W3,
                           const float* __restrict__ b_s, float* __restrict__ Psig) {
    int n = blockIdx.x, v = blockIdx.y;
    int u = threadIdx.x; // 128
    __shared__ float m[FD_];
    __shared__ float lhs[T_];
    const float* xb = xT + (long)n * T_ * V_ * FD_;
    if (u < FD_) {
        float s = 0.f;
        for (int t = 0; t < T_; ++t) s += xb[((long)t * V_ + v) * FD_ + u] * W1[t];
        m[u] = s;
    }
    __syncthreads();
    if (u < T_) {
        float s = 0.f;
        for (int f = 0; f < FD_; ++f) s += m[f] * W2[f * T_ + u];
        lhs[u] = s;
    }
    __syncthreads();
    float r[T_];
    for (int t = 0; t < T_; ++t) {
        float s = 0.f;
        for (int f = 0; f < FD_; ++f) s += xb[((long)t * V_ + u) * FD_ + f] * W3[f];
        r[t] = s;
    }
    float p = 0.f;
    for (int t = 0; t < T_; ++t) p += lhs[t] * r[t];
    p += b_s[v * V_ + u];
    Psig[((long)n * V_ + v) * V_ + u] = 1.f / (1.f + expf(-p));
}

// softmax over axis=1 (rows i) for each (n, column j)
__global__ void k_softmax_col(const float* __restrict__ Sv, float* __restrict__ SAtt) {
    int n = blockIdx.x, j = blockIdx.y;
    int i = threadIdx.x; // 128
    __shared__ float red[128];
    float v = Sv[((long)n * V_ + i) * V_ + j];
    red[i] = v; __syncthreads();
    for (int s = 64; s > 0; s >>= 1) { if (i < s) red[i] = fmaxf(red[i], red[i + s]); __syncthreads(); }
    float mx = red[0]; __syncthreads();
    float e = expf(v - mx);
    red[i] = e; __syncthreads();
    for (int s = 64; s > 0; s >>= 1) { if (i < s) red[i] += red[i + s]; __syncthreads(); }
    SAtt[((long)n * V_ + i) * V_ + j] = e / red[0];
}

// graph learning: tmpS[n,t,i,j] = exp(dot/FD), DD[i,j] = sum_f diff_mean^2 (deterministic)
__global__ void k_graph(const float* __restrict__ x, const float* __restrict__ a,
                        float* __restrict__ tmpS, float* __restrict__ DD) {
    int i = blockIdx.x, j = blockIdx.y;
    int tid = threadIdx.x; // 64
    {   // pass 1: tid = (n,t)
        int n = tid >> 3, t = tid & 7;
        const float* xi = x + (((long)(n * T_ + t) * V_) + i) * FD_;
        const float* xj = x + (((long)(n * T_ + t) * V_) + j) * FD_;
        float dot = 0.f;
        for (int f = 0; f < FD_; ++f) dot += fabsf(xi[f] - xj[f]) * a[f];
        tmpS[(((long)(n * T_ + t) * V_) + i) * V_ + j] = expf(dot / (float)FD_);
    }
    // pass 2: tid = f
    float dm = 0.f;
    for (int nt = 0; nt < NT_; ++nt) {
        const float* xb = x + ((long)nt * V_) * FD_;
        dm += fabsf(xb[(long)i * FD_ + tid] - xb[(long)j * FD_ + tid]);
    }
    dm *= (1.f / (float)NT_);
    __shared__ float red[64];
    red[tid] = dm * dm; __syncthreads();
    for (int s = 32; s > 0; s >>= 1) { if (tid < s) red[tid] += red[tid + s]; __syncthreads(); }
    if (tid == 0) DD[i * V_ + j] = red[0];
}

// column-normalize + 0.7/0.3 eye mix (in place)
__global__ void k_norm_S(float* __restrict__ S) {
    int nt = blockIdx.x, j = blockIdx.y;
    int i = threadIdx.x; // 128
    __shared__ float red[128];
    float* Sb = S + (long)nt * V_ * V_;
    float v = Sb[(long)i * V_ + j];
    red[i] = v; __syncthreads();
    for (int s = 64; s > 0; s >>= 1) { if (i < s) red[i] += red[i + s]; __syncthreads(); }
    float cs = red[0];
    Sb[(long)i * V_ + j] = 0.7f * v / (cs + 1e-8f) + ((i == j) ? 0.3f : 0.f);
}

// losses: two-stage deterministic reduction over (t,i,j)
__global__ void k_loss_part(const float* __restrict__ S, const float* __restrict__ DD,
                            float* __restrict__ part) {
    int idx = blockIdx.x * 256 + threadIdx.x; // T*V*V = 131072
    int t = idx >> 14;
    int rem = idx & 16383;
    int i = rem >> 7, j = rem & 127;
    float sm = 0.f;
    for (int n = 0; n < N_; ++n) sm += S[((((long)n * T_ + t) * V_) + i) * V_ + j];
    sm *= (1.f / N_);
    float p1 = DD[rem] * sm, p2 = sm * sm;
    __shared__ float s1[256], s2[256];
    s1[threadIdx.x] = p1; s2[threadIdx.x] = p2; __syncthreads();
    for (int s = 128; s > 0; s >>= 1) {
        if (threadIdx.x < s) { s1[threadIdx.x] += s1[threadIdx.x + s]; s2[threadIdx.x] += s2[threadIdx.x + s]; }
        __syncthreads();
    }
    if (threadIdx.x == 0) { part[blockIdx.x * 2] = s1[0]; part[blockIdx.x * 2 + 1] = s2[0]; }
}
__global__ void k_loss_final(const float* __restrict__ part, float* __restrict__ out01) {
    int tid = threadIdx.x; // 256, over 512 partials
    __shared__ float s1[256], s2[256];
    s1[tid] = part[tid * 2] + part[(tid + 256) * 2];
    s2[tid] = part[tid * 2 + 1] + part[(tid + 256) * 2 + 1];
    __syncthreads();
    for (int s = 128; s > 0; s >>= 1) {
        if (tid < s) { s1[tid] += s1[tid + s]; s2[tid] += s2[tid + s]; }
        __syncthreads();
    }
    if (tid == 0) { out01[0] = ALPHA_ * s1[0]; out01[1] = ALPHA_ * s2[0]; }
}

// symmetrize (min) + Laplacian: L[i,j] = deg_i*eye - Ssym - eye
__global__ void k_sym_L(const float* __restrict__ S, float* __restrict__ L) {
    int nt = blockIdx.x, i = blockIdx.y;
    int j = threadIdx.x; // 128
    const float* Sb = S + (long)nt * V_ * V_;
    float s = fminf(Sb[(long)i * V_ + j], Sb[(long)j * V_ + i]);
    __shared__ float red[128];
    red[j] = s; __syncthreads();
    for (int k = 64; k > 0; k >>= 1) { if (j < k) red[j] += red[j + k]; __syncthreads(); }
    float deg = red[0];
    L[((long)nt * V_ + i) * V_ + j] = (i == j) ? (deg - s - 1.f) : (-s);
}

// k=0 polynomial term: G[nt,i,f] = Att[n,i,i] * X[nt,i,f]
__global__ void k_g0(const float* __restrict__ X, const float* __restrict__ Att,
                     float* __restrict__ G, int Fin, long total) {
    long idx = (long)blockIdx.x * 256 + threadIdx.x;
    if (idx >= total) return;
    long r = idx / Fin;
    int i = (int)(r % V_);
    int nt = (int)(r / V_);
    int n = nt >> 3;
    G[idx] = Att[((long)n * V_ + i) * V_ + i] * X[idx];
}

__global__ void k_relu(float* __restrict__ p, long total) {
    long idx = (long)blockIdx.x * 256 + threadIdx.x;
    if (idx < total) p[idx] = fmaxf(p[idx], 0.f);
}

// time conv (SAME over T, KT=3) + bias + relu + LayerNorm over channels
__global__ void k_timeconv(const float* __restrict__ XGL, const float* __restrict__ w,
                           const float* __restrict__ bias, float* __restrict__ y) {
    int nt = blockIdx.x, v = blockIdx.y;
    int o = threadIdx.x; // 64
    int n = nt >> 3, t = nt & 7;
    __shared__ float xs[KT_][CHEV_];
    for (int kt = 0; kt < KT_; ++kt) {
        int tt = t + kt - 1;
        xs[kt][o] = (tt >= 0 && tt < T_) ? XGL[(((long)(n * T_ + tt) * V_) + v) * CHEV_ + o] : 0.f;
    }
    __syncthreads();
    float acc = bias[o];
    for (int kt = 0; kt < KT_; ++kt)
        for (int c = 0; c < CHEV_; ++c)
            acc += xs[kt][c] * w[(o * CHEV_ + c) * KT_ + kt];
    acc = fmaxf(acc, 0.f);
    __shared__ float red[64];
    red[o] = acc; __syncthreads();
    for (int s = 32; s > 0; s >>= 1) { if (o < s) red[o] += red[o + s]; __syncthreads(); }
    float mu = red[0] / 64.f;
    __syncthreads();
    float d = acc - mu;
    red[o] = d * d; __syncthreads();
    for (int s = 32; s > 0; s >>= 1) { if (o < s) red[o] += red[o + s]; __syncthreads(); }
    float var = red[0] / 64.f;
    y[(((long)nt * V_) + v) * TF_ + o] = d * rsqrtf(var + 1e-5f);
}

// =============================== host launch ===============================
extern "C" void kernel_launch(void* const* d_in, const int* in_sizes, int n_in,
                              void* d_out, int out_size, void* d_ws, size_t ws_size,
                              hipStream_t stream) {
    const float* x      = (const float*)d_in[0];
    const float* a      = (const float*)d_in[1];
    const float* W1     = (const float*)d_in[2];
    const float* W2     = (const float*)d_in[3];
    const float* W3     = (const float*)d_in[4];
    const float* b_s    = (const float*)d_in[5];
    const float* V_s    = (const float*)d_in[6];
    const float* U1     = (const float*)d_in[7];
    const float* U2     = (const float*)d_in[8];
    const float* U3     = (const float*)d_in[9];
    const float* b_e    = (const float*)d_in[10];
    const float* V_e    = (const float*)d_in[11];
    const float* Theta1 = (const float*)d_in[12];
    const float* Theta2 = (const float*)d_in[13];
    const float* conv_w = (const float*)d_in[14];
    const float* conv_b = (const float*)d_in[15];
    float* y = (float*)d_out;
    float* out01 = y + YSZ_;

    float* ws   = (float*)d_ws;
    float* TAtt = ws;                       // 512
    float* xT   = TAtt + N_ * T_ * T_;      // 524288
    float* Psig = xT + (long)N_ * T_ * V_ * FD_;   // 131072
    float* Sv   = Psig + (long)N_ * V_ * V_;       // 131072
    float* SAtt = Sv + (long)N_ * V_ * V_;         // 131072
    float* S    = SAtt + (long)N_ * V_ * V_;       // 1048576
    float* DD   = S + (long)NT_ * V_ * V_;         // 16384
    float* L    = DD + (long)V_ * V_;              // 1048576
    float* G    = L + (long)NT_ * V_ * V_;         // 2097152
    float* H1   = G + (long)NT_ * V_ * C1_;        // 2097152
    float* XGL  = H1 + (long)NT_ * V_ * C1_;       // 524288
    float* lossP = XGL + (long)NT_ * V_ * CHEV_;   // 1024

    // ---- temporal attention ----
    k_tatt<<<N_, 128, 0, stream>>>(x, U1, U2, U3, b_e, V_e, TAtt);
    k_xtatt<<<(N_ * T_ * V_ * FD_) / 256, 256, 0, stream>>>(x, TAtt, xT);

    // ---- spatial attention ----
    k_satt_pre<<<dim3(N_, V_), 128, 0, stream>>>(xT, W1, W2, W3, b_s, Psig);
    wmma_gemm_batched<V_, V_><<<dim3(N_, V_ / 16, V_ / 64), 32, 0, stream>>>(
        V_s, Psig, Sv, 0, (long)V_ * V_, (long)V_ * V_, 1);
    k_softmax_col<<<dim3(N_, V_), 128, 0, stream>>>(Sv, SAtt);

    // ---- graph learning + losses ----
    k_graph<<<dim3(V_, V_), 64, 0, stream>>>(x, a, S, DD);
    k_norm_S<<<dim3(NT_, V_), 128, 0, stream>>>(S);
    k_loss_part<<<(T_ * V_ * V_) / 256, 256, 0, stream>>>(S, DD, lossP);
    k_loss_final<<<1, 256, 0, stream>>>(lossP, out01);

    // ---- shared symmetrized Laplacian ----
    k_sym_L<<<dim3(NT_, V_), 128, 0, stream>>>(S, L);

    // ---- Cheb layer 1: x(V,64) -> H1(V,256) ----
    {
        long totG = (long)NT_ * V_ * FD_;
        k_g0<<<(int)((totG + 255) / 256), 256, 0, stream>>>(x, SAtt, G, FD_, totG);
        wmma_gemm_batched<FD_, C1_><<<dim3(NT_, V_ / 16, C1_ / 64), 32, 0, stream>>>(
            G, Theta1 + 0 * FD_ * C1_, H1, (long)V_ * FD_, 0, (long)V_ * C1_, 1);
        for (int k = 1; k < 3; ++k) {
            cheb_agg_wmma<FD_><<<dim3(NT_, V_ / 16, FD_ / 64), 32, 0, stream>>>(L, SAtt, x, G, k);
            wmma_gemm_batched<FD_, C1_><<<dim3(NT_, V_ / 16, C1_ / 64), 32, 0, stream>>>(
                G, Theta1 + (long)k * FD_ * C1_, H1, (long)V_ * FD_, 0, (long)V_ * C1_, 0);
        }
        long totH = (long)NT_ * V_ * C1_;
        k_relu<<<(int)((totH + 255) / 256), 256, 0, stream>>>(H1, totH);
    }

    // ---- Cheb layer 2: H1(V,256) -> XGL(V,64) ----
    {
        long totG = (long)NT_ * V_ * C1_;
        k_g0<<<(int)((totG + 255) / 256), 256, 0, stream>>>(H1, SAtt, G, C1_, totG);
        wmma_gemm_batched<C1_, CHEV_><<<dim3(NT_, V_ / 16, CHEV_ / 64), 32, 0, stream>>>(
            G, Theta2 + 0 * C1_ * CHEV_, XGL, (long)V_ * C1_, 0, (long)V_ * CHEV_, 1);
        for (int k = 1; k < 3; ++k) {
            cheb_agg_wmma<C1_><<<dim3(NT_, V_ / 16, C1_ / 64), 32, 0, stream>>>(L, SAtt, H1, G, k);
            wmma_gemm_batched<C1_, CHEV_><<<dim3(NT_, V_ / 16, CHEV_ / 64), 32, 0, stream>>>(
                G, Theta2 + (long)k * C1_ * CHEV_, XGL, (long)V_ * C1_, 0, (long)V_ * CHEV_, 0);
        }
        long totX = (long)NT_ * V_ * CHEV_;
        k_relu<<<(int)((totX + 255) / 256), 256, 0, stream>>>(XGL, totX);
    }

    // ---- time conv + LayerNorm -> y ----
    k_timeconv<<<dim3(NT_, V_), 64, 0, stream>>>(XGL, conv_w, conv_b, y);
}
